// PatchedMixtralAttention_43533788512803
// MI455X (gfx1250) — matrix operand order repaired and checked by reading, not compiled, not run on hardware
//
#include <hip/hip_runtime.h>
#include <hip/hip_bf16.h>

typedef __attribute__((ext_vector_type(16))) _Float16 v16h;
typedef __attribute__((ext_vector_type(8)))  _Float16 v8h;
typedef __attribute__((ext_vector_type(8)))  float    v8f;

#define N_HEADS 32
#define N_KV 8
#define HEAD_DIM 128
#define Q_LEN 1024
#define HISTORY 1024
#define KV_LEN 2048
#define WINDOW 1024
#define BATCH 4
#define T_TOK (BATCH * Q_LEN)                 // 4096
#define HID (N_HEADS * HEAD_DIM)              // 4096
#define KVH (N_KV * HEAD_DIM)                 // 1024
#define QKV_N (HID + 2 * KVH)                 // 6144

// Build a v16h WMMA operand from two contiguous 16-byte runs (forces ds/global b128).
__device__ __forceinline__ v16h ld_frag16(const _Float16* p0, const _Float16* p1) {
    v8h a = *reinterpret_cast<const v8h*>(p0);
    v8h b = *reinterpret_cast<const v8h*>(p1);
    return __builtin_shufflevector(a, b, 0, 1, 2, 3, 4, 5, 6, 7,
                                         8, 9, 10, 11, 12, 13, 14, 15);
}

// ---------------------------------------------------------------------------
// Weight pack: wq|wk|wv -> f16, stored TRANSPOSED as [QKV_N, HID] (N-major).
// ---------------------------------------------------------------------------
__global__ void pack_qkv_wT(const float* __restrict__ wq, const float* __restrict__ wk,
                            const float* __restrict__ wv, _Float16* __restrict__ dst) {
    long i = (long)blockIdx.x * blockDim.x + threadIdx.x;
    if (i >= (long)HID * QKV_N) return;
    int row = (int)(i % HID);     // K index
    int n   = (int)(i / HID);     // output column
    float v;
    if (n < HID)            v = wq[(long)row * HID + n];
    else if (n < HID + KVH) v = wk[(long)row * KVH + (n - HID)];
    else                    v = wv[(long)row * KVH + (n - HID - KVH)];
    dst[i] = (_Float16)v;
}

// Generic transpose-convert: src [R, N] f32 row-major -> dst [N, R] f16.
__global__ void transpose_f32_to_f16(const float* __restrict__ src, _Float16* __restrict__ dst,
                                     int R, int Ncols) {
    long i = (long)blockIdx.x * blockDim.x + threadIdx.x;
    if (i >= (long)R * Ncols) return;
    int r = (int)(i % R);
    int n = (int)(i / R);
    dst[i] = (_Float16)src[(long)r * Ncols + n];
}

__global__ void f32_to_f16(const float* __restrict__ src, _Float16* __restrict__ dst, long n) {
    long i = (long)blockIdx.x * blockDim.x + threadIdx.x;
    if (i < n) dst[i] = (_Float16)src[i];
}

// ---------------------------------------------------------------------------
// Tiled f16 GEMM, f32 accumulate: C[M,N] = A[M,K] @ B[K,N], B given TRANSPOSED
// as BT[N,K]. 256 threads = 8 waves; block tile 128x128x32; wave tile 32x64.
// Two-stage pipeline: global fetch of tile k+1 overlaps WMMA compute on tile k.
// ---------------------------------------------------------------------------
#define BM 128
#define BN 128
#define BK 32

__global__ __launch_bounds__(256)
void gemm_f16_bt(const _Float16* __restrict__ A, const _Float16* __restrict__ BT,
                 void* __restrict__ C, int M, int N, int K, int storeHalf)
{
    __shared__ _Float16 As[BM][BK + 8];     // stride 40 halves = 80B (16B aligned)
    __shared__ _Float16 BsT[BN][BK + 8];    // [col][k], stride 80B

    const int tid = threadIdx.x;
    const int lane = tid & 31;
    const int wave = tid >> 5;
    const int m0 = blockIdx.y * BM;
    const int n0 = blockIdx.x * BN;
    const int wm = (wave >> 1) * 32;
    const int wn = (wave & 1) * 64;
    const int l15 = lane & 15;
    const int hiHalf = lane >> 4;

    // per-thread staging slot: rows (row, row+64), 8-half chunk c4
    const int row = tid >> 2;
    const int c4  = (tid & 3) * 8;
    const _Float16* Ap0 = A  + (long)(m0 + row) * K + c4;
    const _Float16* Ap1 = A  + (long)(m0 + row + 64) * K + c4;
    const _Float16* Bp0 = BT + (long)(n0 + row) * K + c4;
    const _Float16* Bp1 = BT + (long)(n0 + row + 64) * K + c4;

    v8f acc[2][4] = {};

    // prefetch tile 0
    uint4 ra0 = *reinterpret_cast<const uint4*>(Ap0);
    uint4 ra1 = *reinterpret_cast<const uint4*>(Ap1);
    uint4 rb0 = *reinterpret_cast<const uint4*>(Bp0);
    uint4 rb1 = *reinterpret_cast<const uint4*>(Bp1);

    for (int kb = 0; kb < K; kb += BK) {
        // stage current tile to LDS
        *reinterpret_cast<uint4*>(&As[row][c4])       = ra0;
        *reinterpret_cast<uint4*>(&As[row + 64][c4])  = ra1;
        *reinterpret_cast<uint4*>(&BsT[row][c4])      = rb0;
        *reinterpret_cast<uint4*>(&BsT[row + 64][c4]) = rb1;
        __syncthreads();

        // issue next tile's global loads (overlap with WMMA below)
        const int kn = kb + BK;
        if (kn < K) {
            ra0 = *reinterpret_cast<const uint4*>(Ap0 + kn);
            ra1 = *reinterpret_cast<const uint4*>(Ap1 + kn);
            rb0 = *reinterpret_cast<const uint4*>(Bp0 + kn);
            rb1 = *reinterpret_cast<const uint4*>(Bp1 + kn);
        }

        // A frags: lane&15 = row, element e -> K = e + 8*hiHalf (+16 for e>=8)
        v16h af[2];
        {
            int ko = hiHalf * 8;
            #pragma unroll
            for (int mi = 0; mi < 2; ++mi) {
                int arow = wm + mi * 16 + l15;
                af[mi] = ld_frag16(&As[arow][ko], &As[arow][16 + ko]);
            }
        }
        // preload all 4 B frags, then burst 8 independent-chain WMMAs
        v16h bf[4];
        {
            int ko = hiHalf * 8;
            #pragma unroll
            for (int ni = 0; ni < 4; ++ni) {
                int col = wn + ni * 16 + l15;
                bf[ni] = ld_frag16(&BsT[col][ko], &BsT[col][ko + 16]);
            }
        }
        #pragma unroll
        for (int ni = 0; ni < 4; ++ni)
            #pragma unroll
            for (int mi = 0; mi < 2; ++mi)
                acc[mi][ni] = __builtin_amdgcn_wmma_f32_16x16x32_f16(
                    false, af[mi], false, bf[ni], (short)0, acc[mi][ni], false, false);
        __syncthreads();
    }

    // C layout: VGPR r <-> row r + 8*hiHalf, col = lane&15
    #pragma unroll
    for (int mi = 0; mi < 2; ++mi) {
        #pragma unroll
        for (int ni = 0; ni < 4; ++ni) {
            int col = n0 + wn + ni * 16 + l15;
            #pragma unroll
            for (int r = 0; r < 8; ++r) {
                int orow = m0 + wm + mi * 16 + r + hiHalf * 8;
                float v = acc[mi][ni][r];
                if (storeHalf) ((_Float16*)C)[(long)orow * N + col] = (_Float16)v;
                else           ((float*)C)[(long)orow * N + col] = v;
            }
        }
    }
}

// ---------------------------------------------------------------------------
// RoPE + repack: qkv[T,6144] -> q[b,32,1024,128], k (rotated) into
// kbuf[b,8,2048,128] at pos history+i, v copied into d-major vbuf[b,8,128,2048].
// ---------------------------------------------------------------------------
__global__ void rope_pack(const _Float16* __restrict__ qkv,
                          _Float16* __restrict__ qo,
                          _Float16* __restrict__ ko,
                          _Float16* __restrict__ vo)
{
    const int PER = 3584;
    long tid = (long)blockIdx.x * blockDim.x + threadIdx.x;
    long t = tid / PER;
    int  r = (int)(tid % PER);
    if (t >= T_TOK) return;
    int b = (int)(t / Q_LEN);
    int i = (int)(t % Q_LEN);
    int pos = HISTORY + i;
    const _Float16* base = qkv + t * QKV_N;

    if (r < 2560) {
        const _Float16* src;
        _Float16* dst;
        int j;
        if (r < 2048) {
            int head = r >> 6; j = r & 63;
            src = base + head * HEAD_DIM;
            dst = qo + (((long)(b * N_HEADS + head) * Q_LEN) + i) * HEAD_DIM;
        } else {
            int rr = r - 2048;
            int head = rr >> 6; j = rr & 63;
            src = base + HID + head * HEAD_DIM;
            dst = ko + (((long)(b * N_KV + head) * KV_LEN) + pos) * HEAD_DIM;
        }
        float inv = powf(1000000.0f, -(float)j * (1.0f / 64.0f));
        float fr = (float)pos * inv;
        float c = cosf(fr), sn = sinf(fr);
        float x0 = (float)src[j], x1 = (float)src[j + 64];
        dst[j]      = (_Float16)(x0 * c - x1 * sn);
        dst[j + 64] = (_Float16)(x1 * c + x0 * sn);
    } else {
        int e = r - 2560;
        int kvh = e >> 7, d0 = e & 127;
        vo[(((long)(b * N_KV + kvh) * HEAD_DIM) + d0) * KV_LEN + pos] =
            base[HID + KVH + e];
    }
}

// ---------------------------------------------------------------------------
// History gather: contiguous paging => slot = b*KV_LEN + pos.
// ---------------------------------------------------------------------------
__global__ void gather_hist(const float* __restrict__ kc, const float* __restrict__ vc,
                            _Float16* __restrict__ ko, _Float16* __restrict__ vo)
{
    long idx = (long)blockIdx.x * blockDim.x + threadIdx.x;
    if (idx >= (long)BATCH * HISTORY * N_KV * HEAD_DIM) return;
    int d   = (int)(idx & 127);
    int kvh = (int)((idx >> 7) & 7);
    int pos = (int)((idx >> 10) & 1023);
    int b   = (int)(idx >> 20);
    long slot = (long)b * KV_LEN + pos;
    long src = (slot * N_KV + kvh) * HEAD_DIM + d;
    ko[(((long)(b * N_KV + kvh) * KV_LEN) + pos) * HEAD_DIM + d] = (_Float16)kc[src];
    vo[(((long)(b * N_KV + kvh) * HEAD_DIM) + d) * KV_LEN + pos] = (_Float16)vc[src];
}

// ---------------------------------------------------------------------------
// Flash attention: grid (q_len/128, 32 heads, 4 batch), 256 threads = 8 waves.
// ---------------------------------------------------------------------------
__global__ __launch_bounds__(256)
void flash_attn(const _Float16* __restrict__ q,     // [B,32,1024,128]
                const _Float16* __restrict__ kbuf,  // [B,8,2048,128]
                const _Float16* __restrict__ vbuf,  // [B,8,128,2048] d-major
                _Float16* __restrict__ attn_out)    // [T,4096]
{
    __shared__ _Float16 Ks[64][HEAD_DIM + 8];   // [key][d]   stride 272B
    __shared__ _Float16 Vs[HEAD_DIM][64 + 8];   // [d][key]   stride 144B
    __shared__ _Float16 Ps[8][16][64 + 8];      // per-wave P tile

    const int tid = threadIdx.x;
    const int lane = tid & 31;
    const int wave = tid >> 5;
    const int l15 = lane & 15;
    const int hiHalf = lane >> 4;
    const int q0 = blockIdx.x * 128;
    const int h  = blockIdx.y;
    const int b  = blockIdx.z;
    const int kvh = h / (N_HEADS / N_KV);

    // Q fragments (16 rows x 128 dims = 4 A-frags), vector loads from global.
    v16h aq[4];
    {
        const _Float16* qp =
            q + (((long)(b * N_HEADS + h) * Q_LEN) + q0 + wave * 16 + l15) * HEAD_DIM;
        int ko = hiHalf * 8;
        #pragma unroll
        for (int dc = 0; dc < 4; ++dc)
            aq[dc] = ld_frag16(qp + dc * 32 + ko, qp + dc * 32 + 16 + ko);
    }

    v8f o[8] = {};
    float row_max[8], row_sum[8];
    #pragma unroll
    for (int r = 0; r < 8; ++r) { row_max[r] = -1e30f; row_sum[r] = 0.0f; }

    const int qpos_lo = HISTORY + q0;
    const int qpos_hi = qpos_lo + 127;
    int klo = qpos_lo - WINDOW + 1; if (klo < 0) klo = 0;
    const int t0 = klo >> 6;
    const int t1 = qpos_hi >> 6;
    const float scale = 0.08838834764831845f;  // 1/sqrt(128)
    const long kv_base = (long)(b * N_KV + kvh);

    for (int kt = t0; kt <= t1; ++kt) {
        const int kbase = kt * 64;
        __syncthreads();  // protect previous iteration's Vs/Ps reads

        #pragma unroll
        for (int l = 0; l < 4; ++l) {
            int idx = tid + l * 256;
            int krow = idx >> 4, kc4 = idx & 15;
            *reinterpret_cast<uint4*>(&Ks[krow][kc4 * 8]) =
                *reinterpret_cast<const uint4*>(
                    kbuf + (kv_base * KV_LEN + kbase + krow) * HEAD_DIM + kc4 * 8);
        }
        #pragma unroll
        for (int l = 0; l < 4; ++l) {
            int idx = tid + l * 256;
            int vrow = idx >> 3, vc4 = idx & 7;
            *reinterpret_cast<uint4*>(&Vs[vrow][vc4 * 8]) =
                *reinterpret_cast<const uint4*>(
                    vbuf + (kv_base * HEAD_DIM + vrow) * KV_LEN + kbase + vc4 * 8);
        }
        __syncthreads();

        // S = Q K^T: dc outer so the 4 s[nt] chains stay independent per burst.
        v8f s[4] = {};
        #pragma unroll
        for (int dc = 0; dc < 4; ++dc) {
            v16h bk[4];
            int ko = hiHalf * 8;
            #pragma unroll
            for (int nt = 0; nt < 4; ++nt) {
                int key = nt * 16 + l15;
                bk[nt] = ld_frag16(&Ks[key][dc * 32 + ko], &Ks[key][dc * 32 + ko + 16]);
            }
            #pragma unroll
            for (int nt = 0; nt < 4; ++nt)
                s[nt] = __builtin_amdgcn_wmma_f32_16x16x32_f16(
                    false, aq[dc], false, bk[nt], (short)0, s[nt], false, false);
        }

        // scale + causal/sliding-window mask + tile row max
        float mloc[8];
        #pragma unroll
        for (int r = 0; r < 8; ++r) mloc[r] = -1e30f;
        #pragma unroll
        for (int nt = 0; nt < 4; ++nt) {
            int kpos = kbase + nt * 16 + l15;
            #pragma unroll
            for (int r = 0; r < 8; ++r) {
                int qpos = qpos_lo + wave * 16 + r + hiHalf * 8;
                float v = s[nt][r] * scale;
                bool ok = (kpos <= qpos) && ((qpos - kpos) < WINDOW);
                v = ok ? v : -1e30f;
                s[nt][r] = v;
                mloc[r] = fmaxf(mloc[r], v);
            }
        }
        #pragma unroll
        for (int r = 0; r < 8; ++r) {
            #pragma unroll
            for (int off = 8; off >= 1; off >>= 1)
                mloc[r] = fmaxf(mloc[r], __shfl_xor(mloc[r], off, 16));
        }

        float corr[8];
        #pragma unroll
        for (int r = 0; r < 8; ++r) {
            float mnew = fmaxf(row_max[r], mloc[r]);
            corr[r] = __expf(row_max[r] - mnew);
            row_max[r] = mnew;
        }

        float rs[8];
        #pragma unroll
        for (int r = 0; r < 8; ++r) rs[r] = 0.0f;
        #pragma unroll
        for (int nt = 0; nt < 4; ++nt) {
            #pragma unroll
            for (int r = 0; r < 8; ++r) {
                float v = s[nt][r];
                float p = (v > -1e29f) ? __expf(v - row_max[r]) : 0.0f;
                s[nt][r] = p;
                rs[r] += p;
            }
        }
        #pragma unroll
        for (int r = 0; r < 8; ++r) {
            #pragma unroll
            for (int off = 8; off >= 1; off >>= 1)
                rs[r] += __shfl_xor(rs[r], off, 16);
            row_sum[r] = row_sum[r] * corr[r] + rs[r];
            #pragma unroll
            for (int dt = 0; dt < 8; ++dt) o[dt][r] *= corr[r];
        }

        // stage P to LDS to re-enter A-fragment layout
        #pragma unroll
        for (int nt = 0; nt < 4; ++nt) {
            #pragma unroll
            for (int r = 0; r < 8; ++r)
                Ps[wave][r + hiHalf * 8][nt * 16 + l15] = (_Float16)s[nt][r];
        }
        __syncthreads();

        // O += P V: kc outer; dt bursts of 4 keep chains independent.
        v16h ap[2];
        {
            int ko = hiHalf * 8;
            #pragma unroll
            for (int kc = 0; kc < 2; ++kc)
                ap[kc] = ld_frag16(&Ps[wave][l15][kc * 32 + ko],
                                   &Ps[wave][l15][kc * 32 + 16 + ko]);
        }
        #pragma unroll
        for (int kc = 0; kc < 2; ++kc) {
            int ko = hiHalf * 8;
            #pragma unroll
            for (int g = 0; g < 2; ++g) {
                v16h bv[4];
                #pragma unroll
                for (int j = 0; j < 4; ++j) {
                    int dcol = (g * 4 + j) * 16 + l15;
                    bv[j] = ld_frag16(&Vs[dcol][kc * 32 + ko], &Vs[dcol][kc * 32 + ko + 16]);
                }
                #pragma unroll
                for (int j = 0; j < 4; ++j)
                    o[g * 4 + j] = __builtin_amdgcn_wmma_f32_16x16x32_f16(
                        false, ap[kc], false, bv[j], (short)0, o[g * 4 + j], false, false);
            }
        }
    }

    // normalize and store [T, 4096] f16
    #pragma unroll
    for (int dt = 0; dt < 8; ++dt) {
        int col = h * HEAD_DIM + dt * 16 + l15;
        #pragma unroll
        for (int r = 0; r < 8; ++r) {
            int qrow = q0 + wave * 16 + r + hiHalf * 8;
            long t = (long)b * Q_LEN + qrow;
            float v = o[dt][r] / row_sum[r];
            attn_out[t * HID + col] = (_Float16)v;
        }
    }
}

// ---------------------------------------------------------------------------
// Launch
// ---------------------------------------------------------------------------
extern "C" void kernel_launch(void* const* d_in, const int* in_sizes, int n_in,
                              void* d_out, int out_size, void* d_ws, size_t ws_size,
                              hipStream_t stream)
{
    (void)in_sizes; (void)n_in; (void)out_size; (void)ws_size;
    const float* hidden = (const float*)d_in[0];
    const float* wq     = (const float*)d_in[1];
    const float* wk     = (const float*)d_in[2];
    const float* wv     = (const float*)d_in[3];
    const float* wo     = (const float*)d_in[4];
    const float* kcache = (const float*)d_in[5];
    const float* vcache = (const float*)d_in[6];
    float* out = (float*)d_out;

    char* ws = (char*)d_ws;
    size_t off = 0;
    auto alloc = [&](size_t bytes) {
        char* p = ws + off;
        off += (bytes + 255) & ~(size_t)255;
        return p;
    };
    _Float16* hidden_h = (_Float16*)alloc((size_t)T_TOK * HID * 2);            // 32 MB
    _Float16* wqkvT_h  = (_Float16*)alloc((size_t)HID * QKV_N * 2);            // 48 MB
    _Float16* qkv_h    = (_Float16*)alloc((size_t)T_TOK * QKV_N * 2);          // 48 MB
    _Float16* q_h      = (_Float16*)alloc((size_t)T_TOK * HID * 2);            // 32 MB
    _Float16* k_h      = (_Float16*)alloc((size_t)BATCH * N_KV * KV_LEN * HEAD_DIM * 2); // 16 MB
    _Float16* v_h      = (_Float16*)alloc((size_t)BATCH * N_KV * KV_LEN * HEAD_DIM * 2); // 16 MB
    _Float16* woT_h  = wqkvT_h;  // alias: wqkvT consumed by QKV GEMM before step 6
    _Float16* attn_h = qkv_h;    // alias: qkv consumed by rope before step 7

    // 1. pack QKV weights to f16, transposed [QKV_N, HID]
    { long n = (long)HID * QKV_N;
      pack_qkv_wT<<<dim3((unsigned)((n + 255) / 256)), 256, 0, stream>>>(wq, wk, wv, wqkvT_h); }
    // 2. hidden -> f16 (row-major A)
    { long n = (long)T_TOK * HID;
      f32_to_f16<<<dim3((unsigned)((n + 255) / 256)), 256, 0, stream>>>(hidden, hidden_h, n); }
    // 3. fused QKV projection -> f16
    gemm_f16_bt<<<dim3(QKV_N / BN, T_TOK / BM), 256, 0, stream>>>(
        hidden_h, wqkvT_h, (void*)qkv_h, T_TOK, QKV_N, HID, 1);
    // 4. RoPE + scatter new K/V
    { long n = (long)T_TOK * 3584;
      rope_pack<<<dim3((unsigned)((n + 255) / 256)), 256, 0, stream>>>(qkv_h, q_h, k_h, v_h); }
    // 5. gather history from (contiguously paged) caches
    { long n = (long)BATCH * HISTORY * N_KV * HEAD_DIM;
      gather_hist<<<dim3((unsigned)((n + 255) / 256)), 256, 0, stream>>>(kcache, vcache, k_h, v_h); }
    // 6. wo -> f16 transposed [HID, HID] (into aliased region, after QKV GEMM)
    { long n = (long)HID * HID;
      transpose_f32_to_f16<<<dim3((unsigned)((n + 255) / 256)), 256, 0, stream>>>(wo, woT_h, HID, HID); }
    // 7. flash attention
    flash_attn<<<dim3(Q_LEN / 128, N_HEADS, BATCH), 256, 0, stream>>>(q_h, k_h, v_h, attn_h);
    // 8. output projection -> f32 d_out
    gemm_f16_bt<<<dim3(HID / BN, T_TOK / BM), 256, 0, stream>>>(
        attn_h, woT_h, (void*)out, T_TOK, HID, HID, 0);
}
